// OneBinDeltaModel_48696339202298
// MI455X (gfx1250) — compile-verified, matrix operand order checked by reading, not run on hardware
//
#include <hip/hip_runtime.h>
#include <hip/hip_bf16.h>
#include <stdint.h>

// ---------------------------------------------------------------------------
// Problem constants
// ---------------------------------------------------------------------------
#define BATCH 1024
#define NCLS  12
#define DN0   2048
#define DN1   1000
#define DN2   500
#define DK3   100
#define DNDIM 3

#define DN1P  1024   // N1 padded to BN multiple (also K of layer 2)
#define DN2P  512    // N2 padded to BN multiple

// GEMM tiling
#define BM 128
#define BN 128
#define BK 32
#define LDSS 40   // padded LDS row stride in halfs (80 B; 16 B chunks stay aligned)

typedef _Float16 v16h __attribute__((ext_vector_type(16)));
typedef _Float16 v8h  __attribute__((ext_vector_type(8)));
typedef float    v8f  __attribute__((ext_vector_type(8)));

union HV { v16h v; v8h h[2]; };

// ---------------------------------------------------------------------------
// CDNA5 async global->LDS copy (16B per lane), tracked by ASYNCcnt.
// Generic __shared__ pointer low 32 bits == wave-relative LDS address
// (ISA 10.2 aperture rule), used as the VDST operand.
// ---------------------------------------------------------------------------
static __device__ __forceinline__ void async_cp16(uint32_t lds_addr,
                                                  const _Float16* gptr) {
    asm volatile("global_load_async_to_lds_b128 %0, %1, off"
                 :: "v"(lds_addr), "v"((uint64_t)(uintptr_t)gptr)
                 : "memory");
}
static __device__ __forceinline__ void wait_async0() {
    asm volatile("s_wait_asynccnt 0x0" ::: "memory");
}

// One K-step: load fragments per ISA 7.12.2 16-bit layout, 8 WMMAs.
static __device__ __forceinline__ void mma_step(
    const _Float16* __restrict__ As, const _Float16* __restrict__ Bs,
    int wm, int wn, int l, int khalf, v8f acc[4][2])
{
    HV afr[4], bfr[2];
    #pragma unroll
    for (int i = 0; i < 4; i++) {
        const _Float16* p = &As[(wm + i * 16 + l) * LDSS + khalf];
        afr[i].h[0] = *(const v8h*)p;
        afr[i].h[1] = *(const v8h*)(p + 16);
    }
    #pragma unroll
    for (int j = 0; j < 2; j++) {
        const _Float16* p = &Bs[(wn + j * 16 + l) * LDSS + khalf];
        bfr[j].h[0] = *(const v8h*)p;
        bfr[j].h[1] = *(const v8h*)(p + 16);
    }
    #pragma unroll
    for (int i = 0; i < 4; i++)
        #pragma unroll
        for (int j = 0; j < 2; j++)
            acc[i][j] = __builtin_amdgcn_wmma_f32_16x16x32_f16(
                false, afr[i].v, false, bfr[j].v,
                (short)0, acc[i][j], false, false);
}

// ---------------------------------------------------------------------------
// Batched (per-class) GEMM on pre-converted, pre-padded f16 operands:
//   C[z][m][n] (f16, ldc) = sum_k A[z][m][k] * B[z][n][k]
// All dims are tile multiples => no guards, no tails. Double-buffered LDS
// staged entirely with async global->LDS DMA (ASYNCcnt).
// ---------------------------------------------------------------------------
__global__ __launch_bounds__(256) void gemm_f16(
    const _Float16* __restrict__ A, long aBatch,
    const _Float16* __restrict__ Bw, long bBatch,
    _Float16* __restrict__ Cout, long cBatch,
    int Kdim, int ldc)
{
    __shared__ __align__(16) _Float16 As[2][BM * LDSS];
    __shared__ __align__(16) _Float16 Bs[2][BN * LDSS];

    const int z    = blockIdx.z;
    const int m0   = blockIdx.y * BM;
    const int n0   = blockIdx.x * BN;
    const int tid  = threadIdx.x;
    const int lane = tid & 31;
    const int w    = tid >> 5;
    const int wm   = (w & 1) * 64;   // 2 waves along M, 64 rows each
    const int wn   = (w >> 1) * 32;  // 4 waves along N, 32 cols each

    const _Float16* Ab = A  + (size_t)z * aBatch;
    const _Float16* Bb = Bw + (size_t)z * bBatch;
    _Float16*       Cb = Cout + (size_t)z * cBatch;

    // staging map: 512 x 16B chunks per tile; thread t copies chunks t, t+256
    const int r0   = tid >> 2;        // row 0..63 (and +64)
    const int part = (tid & 3) * 8;   // halfword offset within 64B row segment

    const _Float16* ga0 = Ab + (size_t)(m0 + r0)      * Kdim + part;
    const _Float16* ga1 = Ab + (size_t)(m0 + r0 + 64) * Kdim + part;
    const _Float16* gb0 = Bb + (size_t)(n0 + r0)      * Kdim + part;
    const _Float16* gb1 = Bb + (size_t)(n0 + r0 + 64) * Kdim + part;

    const uint32_t la0 = (uint32_t)(uintptr_t)&As[0][ r0       * LDSS + part];
    const uint32_t la1 = (uint32_t)(uintptr_t)&As[0][(r0 + 64) * LDSS + part];
    const uint32_t lb0 = (uint32_t)(uintptr_t)&Bs[0][ r0       * LDSS + part];
    const uint32_t lb1 = (uint32_t)(uintptr_t)&Bs[0][(r0 + 64) * LDSS + part];
    const uint32_t bufStrideA = (uint32_t)(BM * LDSS * sizeof(_Float16));
    const uint32_t bufStrideB = (uint32_t)(BN * LDSS * sizeof(_Float16));

    const int l     = lane & 15;
    const int khalf = (lane >> 4) * 8;
    const int nkt   = Kdim / BK;

    v8f acc[4][2] = {};

    // prologue: stage tile 0 into buffer 0
    async_cp16(la0, ga0);
    async_cp16(la1, ga1);
    async_cp16(lb0, gb0);
    async_cp16(lb1, gb1);
    wait_async0();
    __syncthreads();

    for (int kt = 0; kt < nkt; ++kt) {
        const int cur = kt & 1;
        if (kt + 1 < nkt) {
            const uint32_t nb = (uint32_t)(kt + 1) & 1;
            const int koff = (kt + 1) * BK;
            async_cp16(la0 + nb * bufStrideA, ga0 + koff);
            async_cp16(la1 + nb * bufStrideA, ga1 + koff);
            async_cp16(lb0 + nb * bufStrideB, gb0 + koff);
            async_cp16(lb1 + nb * bufStrideB, gb1 + koff);
        }
        mma_step(As[cur], Bs[cur], wm, wn, l, khalf, acc);
        if (kt + 1 < nkt) {
            wait_async0();
            __syncthreads();
        }
    }

    // epilogue: f16 output, no guards (padded buffers)
    const int hi = (lane >> 4) * 8;
    #pragma unroll
    for (int i = 0; i < 4; i++) {
        const int rowbase = m0 + wm + i * 16 + hi;
        #pragma unroll
        for (int j = 0; j < 2; j++) {
            const int col = n0 + wn + j * 16 + l;
            _Float16* cp = Cb + (size_t)rowbase * ldc + col;
            #pragma unroll
            for (int r = 0; r < 8; r++)
                cp[(size_t)r * ldc] = (_Float16)acc[i][j][r];
        }
    }
}

// ---------------------------------------------------------------------------
// f32 -> f16 convert with zero padding: src [C][N][K] -> dst [C][Npad][Kpad]
// ---------------------------------------------------------------------------
__global__ __launch_bounds__(256) void cvt_pad_k(
    const float* __restrict__ src, _Float16* __restrict__ dst,
    int N, int K, int Npad, int Kpad, unsigned total)
{
    const unsigned idx = blockIdx.x * blockDim.x + threadIdx.x;
    if (idx >= total) return;
    const unsigned perC = (unsigned)Npad * (unsigned)Kpad;
    const unsigned c = idx / perC;
    const unsigned rem = idx - c * perC;
    const unsigned n = rem / (unsigned)Kpad;
    const unsigned k = rem - n * (unsigned)Kpad;
    float v = 0.0f;
    if ((int)n < N && (int)k < K)
        v = src[((size_t)c * N + n) * K + k];
    dst[idx] = (_Float16)v;
}

// ---------------------------------------------------------------------------
// Batch-norm stats over the batch axis on f16 activations (f32 accumulate):
// one thread per (class, feature). scale = g*rsqrt(var+eps); shift = b - mu*scale
// ---------------------------------------------------------------------------
__global__ __launch_bounds__(256) void bn_stats_k(
    const _Float16* __restrict__ h, const float* __restrict__ g,
    const float* __restrict__ bias, float* __restrict__ scale,
    float* __restrict__ shift, int N, int ld, int Bn, int total)
{
    const int idx = blockIdx.x * blockDim.x + threadIdx.x;
    if (idx >= total) return;
    const int c = idx / N;
    const int m = idx - c * N;
    const _Float16* p = h + (size_t)c * Bn * ld + m;
    float s = 0.0f, s2 = 0.0f;
    for (int b = 0; b < Bn; b++) {
        float v = (float)p[(size_t)b * ld];
        s  += v;
        s2 += v * v;
    }
    const float invB = 1.0f / (float)Bn;
    const float mu   = s * invB;
    const float var  = s2 * invB - mu * mu;
    const float inv  = rsqrtf(var + 1e-5f);
    const float sc   = g[idx] * inv;
    scale[idx] = sc;
    shift[idx] = bias[idx] - mu * sc;
}

// ---------------------------------------------------------------------------
// In-place normalize + ReLU on the valid region of padded f16 activations.
// ---------------------------------------------------------------------------
__global__ __launch_bounds__(256) void bn_apply_k(
    _Float16* __restrict__ h, const float* __restrict__ scale,
    const float* __restrict__ shift, int N, int ld, int Bn, unsigned total)
{
    const unsigned idx = blockIdx.x * blockDim.x + threadIdx.x;
    if (idx >= total) return;                      // total = C*Bn*N (valid only)
    const unsigned row = idx / (unsigned)N;        // c*Bn + b
    const unsigned m   = idx - row * (unsigned)N;
    const unsigned c   = row / (unsigned)Bn;
    const unsigned cn  = c * (unsigned)N + m;
    const size_t pos   = (size_t)row * ld + m;
    const float v = fmaf((float)h[pos], scale[cn], shift[cn]);
    h[pos] = (_Float16)(v > 0.0f ? v : 0.0f);
}

// ---------------------------------------------------------------------------
// Layer 3 with label gather:
//   out[b][k] = sum_m h2[lab[b]][b][m] * W3[lab[b]][k][m] + b3[lab[b]][k]
// One block per sample; h2 row staged in LDS as f32.
// ---------------------------------------------------------------------------
__global__ __launch_bounds__(128) void head_out_k(
    const _Float16* __restrict__ h2, const int* __restrict__ label,
    const float* __restrict__ W3, const float* __restrict__ b3,
    float* __restrict__ out, int K3, int N2, int ld, int Bn)
{
    __shared__ float row[DN2];
    const int b = blockIdx.x;
    const int c = label[b];
    const _Float16* hp = h2 + ((size_t)c * Bn + b) * ld;
    for (int m = threadIdx.x; m < N2; m += blockDim.x) row[m] = (float)hp[m];
    __syncthreads();
    for (int k = threadIdx.x; k < K3; k += blockDim.x) {
        const float* wrow = W3 + ((size_t)c * K3 + k) * N2;
        float s = 0.0f;
        for (int m = 0; m < N2; m++) s = fmaf(row[m], wrow[m], s);
        out[(size_t)b * K3 + k] = s + b3[c * K3 + k];
    }
}

// ---------------------------------------------------------------------------
// Launch
// ---------------------------------------------------------------------------
extern "C" void kernel_launch(void* const* d_in, const int* in_sizes, int n_in,
                              void* d_out, int out_size, void* d_ws, size_t ws_size,
                              hipStream_t stream)
{
    (void)in_sizes; (void)n_in; (void)out_size; (void)ws_size;

    const float* x     = (const float*)d_in[0];
    const int*   label = (const int*)d_in[1];

    // f16 workspace carve-out
    _Float16* xh  = (_Float16*)d_ws;                          // [B][N0]
    _Float16* w1h = xh  + (size_t)BATCH * DN0;                // [C][N1P][N0]
    _Float16* w2h = w1h + (size_t)NCLS * DN1P * DN0;          // [C][N2P][N1P]
    _Float16* h1h = w2h + (size_t)NCLS * DN2P * DN1P;         // [C][B][N1P]
    _Float16* h2h = h1h + (size_t)NCLS * BATCH * DN1P;        // [C][B][N2P]
    float*    sc1 = (float*)(h2h + (size_t)NCLS * BATCH * DN2P);
    float*    sh1 = sc1 + (size_t)NCLS * DN1;
    float*    sc2 = sh1 + (size_t)NCLS * DN1;
    float*    sh2 = sc2 + (size_t)NCLS * DN2;

    // convert x once (shared by both heads)
    {
        const unsigned tot = (unsigned)BATCH * DN0;
        cvt_pad_k<<<(tot + 255u) / 256u, 256, 0, stream>>>(
            x, xh, BATCH, DN0, BATCH, DN0, tot);
    }

    for (int head = 0; head < 2; ++head) {
        const int base = (head == 0) ? 2 : 10;
        const float* W1 = (const float*)d_in[base + 0];
        const float* g1 = (const float*)d_in[base + 1];
        const float* b1 = (const float*)d_in[base + 2];
        const float* W2 = (const float*)d_in[base + 3];
        const float* g2 = (const float*)d_in[base + 4];
        const float* b2 = (const float*)d_in[base + 5];
        const float* W3 = (const float*)d_in[base + 6];
        const float* b3 = (const float*)d_in[base + 7];
        const int K3 = (head == 0) ? DK3 : DNDIM;
        float* out = (float*)d_out + ((head == 0) ? 0 : (size_t)BATCH * DK3);

        // weight conversion + padding
        {
            const unsigned t1 = (unsigned)NCLS * DN1P * DN0;
            cvt_pad_k<<<(t1 + 255u) / 256u, 256, 0, stream>>>(
                W1, w1h, DN1, DN0, DN1P, DN0, t1);
            const unsigned t2 = (unsigned)NCLS * DN2P * DN1P;
            cvt_pad_k<<<(t2 + 255u) / 256u, 256, 0, stream>>>(
                W2, w2h, DN2, DN1, DN2P, DN1P, t2);
        }

        // layer 1: h1h[c] = xh * w1h[c]^T   (M=1024, Npad=1024, K=2048)
        dim3 grid1(DN1P / BN, BATCH / BM, NCLS);
        gemm_f16<<<grid1, 256, 0, stream>>>(
            xh, 0L, w1h, (long)DN1P * DN0, h1h, (long)BATCH * DN1P,
            DN0, DN1P);

        const int cn1 = NCLS * DN1;
        bn_stats_k<<<(cn1 + 255) / 256, 256, 0, stream>>>(
            h1h, g1, b1, sc1, sh1, DN1, DN1P, BATCH, cn1);
        const unsigned tot1 = (unsigned)NCLS * BATCH * DN1;
        bn_apply_k<<<(tot1 + 255u) / 256u, 256, 0, stream>>>(
            h1h, sc1, sh1, DN1, DN1P, BATCH, tot1);

        // layer 2: h2h[c] = h1h[c] * w2h[c]^T  (M=1024, Npad=512, Kpad=1024)
        dim3 grid2(DN2P / BN, BATCH / BM, NCLS);
        gemm_f16<<<grid2, 256, 0, stream>>>(
            h1h, (long)BATCH * DN1P, w2h, (long)DN2P * DN1P, h2h,
            (long)BATCH * DN2P, DN1P, DN2P);

        const int cn2 = NCLS * DN2;
        bn_stats_k<<<(cn2 + 255) / 256, 256, 0, stream>>>(
            h2h, g2, b2, sc2, sh2, DN2, DN2P, BATCH, cn2);
        const unsigned tot2 = (unsigned)NCLS * BATCH * DN2;
        bn_apply_k<<<(tot2 + 255u) / 256u, 256, 0, stream>>>(
            h2h, sc2, sh2, DN2, DN2P, BATCH, tot2);

        // layer 3 + gather straight into d_out
        head_out_k<<<BATCH, 128, 0, stream>>>(
            h2h, label, W3, b3, out, K3, DN2, DN2P, BATCH);
    }
}
